// MultiHeadAttention_34308198760987
// MI455X (gfx1250) — matrix operand order repaired and checked by reading, not compiled, not run on hardware
//
#include <hip/hip_runtime.h>
#include <hip/hip_bf16.h>

#define DIM      1024
#define N_HEADS  16
#define HEAD_DIM 64
#define BATCH    16
#define SEQ      512

typedef __attribute__((ext_vector_type(16))) __bf16 v16bf;
typedef __attribute__((ext_vector_type(8)))  __bf16 v8bf;
typedef __attribute__((ext_vector_type(4)))  __bf16 v4bf;
typedef __attribute__((ext_vector_type(8)))  float  v8f;
typedef __attribute__((ext_vector_type(4)))  unsigned int u32x4;
typedef __attribute__((ext_vector_type(8)))  int i32x8;
typedef __attribute__((ext_vector_type(4)))  int i32x4;

union Frag { v16bf v; v8bf h[2]; };

#define WMMA_BF16(A, B, C) \
  __builtin_amdgcn_wmma_f32_16x16x32_bf16(false, (A), false, (B), (short)0, (C), false, false)

// ---------------------------------------------------------------------------
// TDM descriptor helper: 2D tile (32 cols x 64 rows of bf16) from a row-major
// [1024 x 1024] bf16 matrix into a compact 64x32 LDS tile.
//   group0: count=1 | lds_addr | global_addr(57b) | type=2
//   group1: data_size=2B, tensor_dim0/1=1024, tile_dim0=32, tile_dim1=64,
//           tensor_dim0_stride=1024
// Toolchain here uses the 6-arg builtin: (g0, g1, g2, g3, x8, cpol).
// ---------------------------------------------------------------------------
__device__ __forceinline__ void tdm_load_w_tile(const __bf16* gptr, unsigned lds_addr) {
  const unsigned long long ga = (unsigned long long)(uintptr_t)gptr;
  u32x4 g0;
  g0[0] = 1u;                                   // count=1, user mode, no gather
  g0[1] = lds_addr;                             // LDS byte address
  g0[2] = (unsigned)ga;                         // global_addr[31:0]
  g0[3] = (unsigned)((ga >> 32) & 0x01FFFFFFu)  // global_addr[56:32]
          | 0x80000000u;                        // type=2 ("image")
  i32x8 g1;
  g1[0] = 0x00010000;      // workgroup_mask=0, data_size=1 (2 bytes)
  g1[1] = 0x04000000;      // tensor_dim0[15:0]=1024 in bits 31:16
  g1[2] = 0x04000000;      // tensor_dim0 hi=0, tensor_dim1[15:0]=1024 in 31:16
  g1[3] = 0x00200000;      // tensor_dim1 hi=0, tile_dim0=32 in bits 31:16
  g1[4] = 64;              // tile_dim1=64, tile_dim2=0
  g1[5] = 1024;            // tensor_dim0_stride lo
  g1[6] = 0;               // stride hi, tensor_dim1_stride lo
  g1[7] = 0;
  i32x4 gz4 = {0, 0, 0, 0};
  i32x8 gz8 = {0, 0, 0, 0, 0, 0, 0, 0};
  __builtin_amdgcn_tensor_load_to_lds(g0, g1, gz4, gz4, gz8, 0);
}

// ---------------------------------------------------------------------------
// Stage 1: f32 -> bf16 conversion (x and the four weight matrices)
// ---------------------------------------------------------------------------
__global__ __launch_bounds__(256) void convert_f32_bf16(const float* __restrict__ src,
                                                        __bf16* __restrict__ dst, int n4) {
  int i = blockIdx.x * blockDim.x + threadIdx.x;
  if (i < n4) {
    float4 f = ((const float4*)src)[i];
    v4bf o;
    o[0] = (__bf16)f.x; o[1] = (__bf16)f.y; o[2] = (__bf16)f.z; o[3] = (__bf16)f.w;
    ((v4bf*)dst)[i] = o;
  }
}

// ---------------------------------------------------------------------------
// Stage 2: QKV projections (x @ W.T) + RoPE. One wave -> 16 tokens x 1 head.
// Weight tiles (64 rows x 32 k) staged into LDS by the Tensor Data Mover,
// double-buffered and overlapped with WMMA; A fragments double-buffered in
// registers. p==0/1 -> Q/K (roped, [B,H,S,Hd]); p==2 -> V (transposed [B,H,Hd,S]).
// ---------------------------------------------------------------------------
__global__ __launch_bounds__(32) void qkv_rope_kernel(
    const __bf16* __restrict__ xb,
    const __bf16* __restrict__ Wq, const __bf16* __restrict__ Wk, const __bf16* __restrict__ Wv,
    __bf16* __restrict__ Qr, __bf16* __restrict__ Kr, __bf16* __restrict__ Vt) {
  __shared__ __bf16 wtile[2][64 * 32];  // two 4 KiB weight tiles (double buffer)

  const int lane = threadIdx.x & 31;
  const int nc = lane & 15;
  const int hi = lane >> 4;
  const int mt = blockIdx.x;   // token tile (16 tokens)
  const int h  = blockIdx.y;   // head
  const int p  = blockIdx.z;   // 0=Q 1=K 2=V
  const __bf16* W = (p == 0) ? Wq : (p == 1) ? Wk : Wv;

  const int m0 = mt * 16;
  const int b  = m0 / SEQ;
  const int s0 = m0 % SEQ;
  const int n0 = h * HEAD_DIM;

  const unsigned lds0 = (unsigned)(uintptr_t)&wtile[0][0];
  const unsigned lds1 = (unsigned)(uintptr_t)&wtile[1][0];
  const __bf16* wrow = W + (size_t)n0 * DIM;  // tile start: row n0, col k0

  v8f c0 = {}, c1 = {}, c2 = {}, c3 = {};

  // A: row (token) = m0 + nc, halves: K = k0 + hi*16 + g*8 + j
  const __bf16* arow = xb + (size_t)(m0 + nc) * DIM + hi * 16;

  // --- prologue: DMA first weight tile, preload first A fragment
  tdm_load_w_tile(wrow, lds0);
  Frag a_cur;
  a_cur.h[0] = *(const v8bf*)(arow);
  a_cur.h[1] = *(const v8bf*)(arow + 8);

  int buf = 0;
  for (int k0 = 0; k0 < DIM; k0 += 32, buf ^= 1) {
    const bool last = (k0 + 32 >= DIM);
    // kick off DMA for next k-block into the other buffer
    if (!last) tdm_load_w_tile(wrow + k0 + 32, buf ? lds0 : lds1);
    // preload next A fragment (overlaps with WMMAs below)
    Frag a_nxt = a_cur;
    if (!last) {
      __builtin_prefetch(arow + k0 + 160, 0, 3);
      a_nxt.h[0] = *(const v8bf*)(arow + k0 + 32);
      a_nxt.h[1] = *(const v8bf*)(arow + k0 + 40);
    }
    // wait for the *current* tile only (next may stay in flight)
    if (!last) __builtin_amdgcn_s_wait_tensorcnt(1);
    else       __builtin_amdgcn_s_wait_tensorcnt(0);
    asm volatile("" ::: "memory");

    const __bf16* lb = &wtile[buf][0];
    Frag b0, b1, b2, b3;
    const __bf16* l0 = lb + (size_t)(nc) * 32 + hi * 16;
    const __bf16* l1 = lb + (size_t)(16 + nc) * 32 + hi * 16;
    const __bf16* l2 = lb + (size_t)(32 + nc) * 32 + hi * 16;
    const __bf16* l3 = lb + (size_t)(48 + nc) * 32 + hi * 16;
    b0.h[0] = *(const v8bf*)l0; b0.h[1] = *(const v8bf*)(l0 + 8);
    b1.h[0] = *(const v8bf*)l1; b1.h[1] = *(const v8bf*)(l1 + 8);
    b2.h[0] = *(const v8bf*)l2; b2.h[1] = *(const v8bf*)(l2 + 8);
    b3.h[0] = *(const v8bf*)l3; b3.h[1] = *(const v8bf*)(l3 + 8);
    c0 = WMMA_BF16(a_cur.v, b0.v, c0);
    c1 = WMMA_BF16(a_cur.v, b1.v, c1);
    c2 = WMMA_BF16(a_cur.v, b2.v, c2);
    c3 = WMMA_BF16(a_cur.v, b3.v, c3);
    a_cur = a_nxt;
    asm volatile("" ::: "memory");  // don't overwrite LDS buffer before reads retire
  }

  if (p < 2) {
    // RoPE: feature d pairs with d+32 -> accumulator pairs (c0,c2) and (c1,c3)
    __bf16* dst = (p == 0) ? Qr : Kr;
    const float kfrq = 0.28782313662425572f;  // ln(10000)/32
    const float inv0 = __expf(-kfrq * (float)(nc));
    const float inv1 = __expf(-kfrq * (float)(16 + nc));
    const size_t hb = ((size_t)b * N_HEADS + h) * SEQ;
#pragma unroll
    for (int r = 0; r < 8; ++r) {
      const int s = s0 + r + hi * 8;
      float sn0, cs0, sn1, cs1;
      __sincosf((float)s * inv0, &sn0, &cs0);
      __sincosf((float)s * inv1, &sn1, &cs1);
      const float q0 = c0[r], q1 = c1[r], q2 = c2[r], q3 = c3[r];
      __bf16* o = dst + (hb + s) * HEAD_DIM + nc;
      o[0]  = (__bf16)(q0 * cs0 - q2 * sn0);
      o[16] = (__bf16)(q1 * cs1 - q3 * sn1);
      o[32] = (__bf16)(q2 * cs0 + q0 * sn0);
      o[48] = (__bf16)(q3 * cs1 + q1 * sn1);
    }
  } else {
    // V transposed: Vt[b,h,d,s]; lane holds fixed d, rows r = consecutive s
    const size_t vb = ((size_t)b * N_HEADS + h) * HEAD_DIM;
#pragma unroll
    for (int t = 0; t < 4; ++t) {
      const v8f cv = (t == 0) ? c0 : (t == 1) ? c1 : (t == 2) ? c2 : c3;
      v8bf pk;
#pragma unroll
      for (int r = 0; r < 8; ++r) pk[r] = (__bf16)cv[r];
      *(v8bf*)(Vt + (vb + t * 16 + nc) * SEQ + s0 + hi * 8) = pk;
    }
  }
}

// ---------------------------------------------------------------------------
// Stage 3: causal flash attention per (b,h,query-tile). One wave per q-tile.
// ---------------------------------------------------------------------------
__device__ inline float redmax16(float v) {
  v = fmaxf(v, __shfl_xor(v, 1, 32));
  v = fmaxf(v, __shfl_xor(v, 2, 32));
  v = fmaxf(v, __shfl_xor(v, 4, 32));
  v = fmaxf(v, __shfl_xor(v, 8, 32));
  return v;
}
__device__ inline float redsum16(float v) {
  v += __shfl_xor(v, 1, 32);
  v += __shfl_xor(v, 2, 32);
  v += __shfl_xor(v, 4, 32);
  v += __shfl_xor(v, 8, 32);
  return v;
}

__global__ __launch_bounds__(32) void attn_kernel(
    const __bf16* __restrict__ Qr, const __bf16* __restrict__ Kr,
    const __bf16* __restrict__ Vt, __bf16* __restrict__ ctx) {
  __shared__ __bf16 plds[16 * 32];  // P tile: 16 queries x 32 keys (bf16)

  const int lane = threadIdx.x & 31;
  const int nc = lane & 15;
  const int hi = lane >> 4;
  const int qt = blockIdx.x;
  const int bh = blockIdx.y;
  const int b = bh / N_HEADS, h = bh % N_HEADS;
  const int q0 = qt * 16;
  const size_t qkb = (size_t)bh * SEQ * HEAD_DIM;
  const size_t vtb = (size_t)bh * HEAD_DIM * SEQ;

  // Q A-fragments (K = 0..31 and 32..63 of head dim)
  const __bf16* qrow = Qr + qkb + (size_t)(q0 + nc) * HEAD_DIM + hi * 16;
  Frag aq0, aq1;
  aq0.h[0] = *(const v8bf*)(qrow);      aq0.h[1] = *(const v8bf*)(qrow + 8);
  aq1.h[0] = *(const v8bf*)(qrow + 32); aq1.h[1] = *(const v8bf*)(qrow + 40);

  float mrun[8], lrun[8];
#pragma unroll
  for (int r = 0; r < 8; ++r) { mrun[r] = -__builtin_inff(); lrun[r] = 0.0f; }
  v8f o0 = {}, o1 = {}, o2 = {}, o3 = {};
  const float scale = 0.125f;  // 1/sqrt(64)

  for (int kb = 0; kb < q0 + 16; kb += 32) {
    // ---- scores: two 16x16 key tiles, Hd reduction split into 2x K32
    v8f s0 = {}, s1 = {};
    const __bf16* kr0 = Kr + qkb + (size_t)(kb + nc) * HEAD_DIM + hi * 16;
    const __bf16* kr1 = kr0 + (size_t)16 * HEAD_DIM;
    __builtin_prefetch(kr0 + 32 * HEAD_DIM, 0, 3);
    Frag bk;
    bk.h[0] = *(const v8bf*)(kr0);      bk.h[1] = *(const v8bf*)(kr0 + 8);
    s0 = WMMA_BF16(aq0.v, bk.v, s0);
    bk.h[0] = *(const v8bf*)(kr0 + 32); bk.h[1] = *(const v8bf*)(kr0 + 40);
    s0 = WMMA_BF16(aq1.v, bk.v, s0);
    bk.h[0] = *(const v8bf*)(kr1);      bk.h[1] = *(const v8bf*)(kr1 + 8);
    s1 = WMMA_BF16(aq0.v, bk.v, s1);
    bk.h[0] = *(const v8bf*)(kr1 + 32); bk.h[1] = *(const v8bf*)(kr1 + 40);
    s1 = WMMA_BF16(aq1.v, bk.v, s1);

    // ---- causal mask + online softmax (rows live across 16-lane halves)
    const int keyc0 = kb + nc;
    const int keyc1 = kb + 16 + nc;
#pragma unroll
    for (int r = 0; r < 8; ++r) {
      const int qi = q0 + r + hi * 8;
      float v0 = (keyc0 > qi) ? -__builtin_inff() : s0[r] * scale;
      float v1 = (keyc1 > qi) ? -__builtin_inff() : s1[r] * scale;
      const float mx = redmax16(fmaxf(v0, v1));
      const float mnew = fmaxf(mrun[r], mx);
      const float alpha = __expf(mrun[r] - mnew);
      mrun[r] = mnew;
      const float p0 = __expf(v0 - mnew);
      const float p1 = __expf(v1 - mnew);
      lrun[r] = lrun[r] * alpha + redsum16(p0 + p1);
      o0[r] *= alpha; o1[r] *= alpha; o2[r] *= alpha; o3[r] *= alpha;
      const int m = r + hi * 8;
      plds[m * 32 + nc]      = (__bf16)p0;
      plds[m * 32 + 16 + nc] = (__bf16)p1;
    }
    asm volatile("s_wait_dscnt 0" ::: "memory");  // cross-lane LDS transpose handoff

    // ---- P as A-fragment (row = query), V as B-fragments (col = d, K = key)
    Frag ap;
    const __bf16* pl = &plds[nc * 32 + hi * 16];
    ap.h[0] = *(const v8bf*)(pl);
    ap.h[1] = *(const v8bf*)(pl + 8);

    const __bf16* vb0 = Vt + vtb + (size_t)(nc)      * SEQ + kb + hi * 16;
    const __bf16* vb1 = Vt + vtb + (size_t)(16 + nc) * SEQ + kb + hi * 16;
    const __bf16* vb2 = Vt + vtb + (size_t)(32 + nc) * SEQ + kb + hi * 16;
    const __bf16* vb3 = Vt + vtb + (size_t)(48 + nc) * SEQ + kb + hi * 16;
    Frag bv;
    bv.h[0] = *(const v8bf*)(vb0); bv.h[1] = *(const v8bf*)(vb0 + 8);
    o0 = WMMA_BF16(ap.v, bv.v, o0);
    bv.h[0] = *(const v8bf*)(vb1); bv.h[1] = *(const v8bf*)(vb1 + 8);
    o1 = WMMA_BF16(ap.v, bv.v, o1);
    bv.h[0] = *(const v8bf*)(vb2); bv.h[1] = *(const v8bf*)(vb2 + 8);
    o2 = WMMA_BF16(ap.v, bv.v, o2);
    bv.h[0] = *(const v8bf*)(vb3); bv.h[1] = *(const v8bf*)(vb3 + 8);
    o3 = WMMA_BF16(ap.v, bv.v, o3);
    asm volatile("" ::: "memory");  // keep LDS reuse ordered across iterations
  }

  // ---- normalize and write context [B,S,D] in bf16
#pragma unroll
  for (int r = 0; r < 8; ++r) {
    const float rcp = 1.0f / lrun[r];
    const int s = q0 + r + hi * 8;
    __bf16* o = ctx + ((size_t)b * SEQ + s) * DIM + h * HEAD_DIM + nc;
    o[0]  = (__bf16)(o0[r] * rcp);
    o[16] = (__bf16)(o1[r] * rcp);
    o[32] = (__bf16)(o2[r] * rcp);
    o[48] = (__bf16)(o3[r] * rcp);
  }
}

// ---------------------------------------------------------------------------
// Stage 4: output projection  out = ctx @ Wo.T  (f32 result), TDM-staged B.
// ---------------------------------------------------------------------------
__global__ __launch_bounds__(32) void out_proj_kernel(
    const __bf16* __restrict__ ctx, const __bf16* __restrict__ Wo,
    float* __restrict__ out) {
  __shared__ __bf16 wtile[2][64 * 32];

  const int lane = threadIdx.x & 31;
  const int nc = lane & 15;
  const int hi = lane >> 4;
  const int m0 = blockIdx.x * 16;
  const int n0 = blockIdx.y * 64;

  const unsigned lds0 = (unsigned)(uintptr_t)&wtile[0][0];
  const unsigned lds1 = (unsigned)(uintptr_t)&wtile[1][0];
  const __bf16* wrow = Wo + (size_t)n0 * DIM;

  v8f c0 = {}, c1 = {}, c2 = {}, c3 = {};
  const __bf16* arow = ctx + (size_t)(m0 + nc) * DIM + hi * 16;

  tdm_load_w_tile(wrow, lds0);
  Frag a_cur;
  a_cur.h[0] = *(const v8bf*)(arow);
  a_cur.h[1] = *(const v8bf*)(arow + 8);

  int buf = 0;
  for (int k0 = 0; k0 < DIM; k0 += 32, buf ^= 1) {
    const bool last = (k0 + 32 >= DIM);
    if (!last) tdm_load_w_tile(wrow + k0 + 32, buf ? lds0 : lds1);
    Frag a_nxt = a_cur;
    if (!last) {
      __builtin_prefetch(arow + k0 + 160, 0, 3);
      a_nxt.h[0] = *(const v8bf*)(arow + k0 + 32);
      a_nxt.h[1] = *(const v8bf*)(arow + k0 + 40);
    }
    if (!last) __builtin_amdgcn_s_wait_tensorcnt(1);
    else       __builtin_amdgcn_s_wait_tensorcnt(0);
    asm volatile("" ::: "memory");

    const __bf16* lb = &wtile[buf][0];
    Frag b0, b1, b2, b3;
    const __bf16* l0 = lb + (size_t)(nc) * 32 + hi * 16;
    const __bf16* l1 = lb + (size_t)(16 + nc) * 32 + hi * 16;
    const __bf16* l2 = lb + (size_t)(32 + nc) * 32 + hi * 16;
    const __bf16* l3 = lb + (size_t)(48 + nc) * 32 + hi * 16;
    b0.h[0] = *(const v8bf*)l0; b0.h[1] = *(const v8bf*)(l0 + 8);
    b1.h[0] = *(const v8bf*)l1; b1.h[1] = *(const v8bf*)(l1 + 8);
    b2.h[0] = *(const v8bf*)l2; b2.h[1] = *(const v8bf*)(l2 + 8);
    b3.h[0] = *(const v8bf*)l3; b3.h[1] = *(const v8bf*)(l3 + 8);
    c0 = WMMA_BF16(a_cur.v, b0.v, c0);
    c1 = WMMA_BF16(a_cur.v, b1.v, c1);
    c2 = WMMA_BF16(a_cur.v, b2.v, c2);
    c3 = WMMA_BF16(a_cur.v, b3.v, c3);
    a_cur = a_nxt;
    asm volatile("" ::: "memory");
  }

#pragma unroll
  for (int r = 0; r < 8; ++r) {
    float* o = out + (size_t)(m0 + r + hi * 8) * DIM + n0 + nc;
    o[0]  = c0[r];
    o[16] = c1[r];
    o[32] = c2[r];
    o[48] = c3[r];
  }
}

// ---------------------------------------------------------------------------
extern "C" void kernel_launch(void* const* d_in, const int* in_sizes, int n_in,
                              void* d_out, int out_size, void* d_ws, size_t ws_size,
                              hipStream_t stream) {
  const float* x  = (const float*)d_in[0];
  const float* Wq = (const float*)d_in[1];
  const float* Wk = (const float*)d_in[2];
  const float* Wv = (const float*)d_in[3];
  const float* Wo = (const float*)d_in[4];

  char* ws = (char*)d_ws;
  const size_t XB   = (size_t)BATCH * SEQ * DIM * 2;                 // 16 MiB
  const size_t WB   = (size_t)DIM * DIM * 2;                         // 2 MiB
  const size_t QKVB = (size_t)BATCH * N_HEADS * SEQ * HEAD_DIM * 2;  // 16 MiB
  __bf16* xb  = (__bf16*)(ws);
  __bf16* Wqb = (__bf16*)(ws + XB);
  __bf16* Wkb = (__bf16*)(ws + XB + WB);
  __bf16* Wvb = (__bf16*)(ws + XB + 2 * WB);
  __bf16* Wob = (__bf16*)(ws + XB + 3 * WB);
  __bf16* Qr  = (__bf16*)(ws + XB + 4 * WB);
  __bf16* Kr  = (__bf16*)(ws + XB + 4 * WB + QKVB);
  __bf16* Vt  = (__bf16*)(ws + XB + 4 * WB + 2 * QKVB);
  __bf16* ctx = (__bf16*)(ws + XB + 4 * WB + 3 * QKVB);

  const int nx4 = BATCH * SEQ * DIM / 4;
  const int nw4 = DIM * DIM / 4;
  convert_f32_bf16<<<nx4 / 256, 256, 0, stream>>>(x, xb, nx4);
  convert_f32_bf16<<<nw4 / 256, 256, 0, stream>>>(Wq, Wqb, nw4);
  convert_f32_bf16<<<nw4 / 256, 256, 0, stream>>>(Wk, Wkb, nw4);
  convert_f32_bf16<<<nw4 / 256, 256, 0, stream>>>(Wv, Wvb, nw4);
  convert_f32_bf16<<<nw4 / 256, 256, 0, stream>>>(Wo, Wob, nw4);

  qkv_rope_kernel<<<dim3(BATCH * SEQ / 16, N_HEADS, 3), 32, 0, stream>>>(
      xb, Wqb, Wkb, Wvb, Qr, Kr, Vt);

  attn_kernel<<<dim3(SEQ / 16, BATCH * N_HEADS), 32, 0, stream>>>(Qr, Kr, Vt, ctx);

  out_proj_kernel<<<dim3(BATCH * SEQ / 16, DIM / 64), 32, 0, stream>>>(
      ctx, Wob, (float*)d_out);
}
